// ConvAttention2d_46334107189729
// MI455X (gfx1250) — matrix operand order repaired and checked by reading, not compiled, not run on hardware
//
#include <hip/hip_runtime.h>
#include <hip/hip_fp16.h>

typedef __attribute__((ext_vector_type(16))) _Float16 v16h;
typedef __attribute__((ext_vector_type(8)))  _Float16 v8h;
typedef __attribute__((ext_vector_type(8)))  float    v8f;

#define DIM_C  384
#define INNER  512
#define HEADS  8
#define DH     64
#define HGT    56
#define WID    56
#define HW     (HGT*WID)      // 3136
#define NBATCH 4
#define NPIX   (NBATCH*HW)    // 12544
#define ATT_SCALE 0.125f      // 64^-0.5

#define V16(lo, hi) __builtin_shufflevector(lo, hi, 0,1,2,3,4,5,6,7,8,9,10,11,12,13,14,15)

// ---------------------------------------------------------------------------
// WMMA GEMM:  C[M x N] = A(f32, MxK row-major) * B(KxN) (+bias)
//   B element addr = k*bRowStride + n + (n/HW)*bBatchExtra
//   C element addr = m*cRowStride + n + (n/HW)*cBatchExtra
// Block tile 128M x 64N, BK=32, 256 threads = 8 waves, one 16M-slice per wave.
// Each wave: 1 A fragment + 4 B fragments -> 4 x v_wmma_f32_16x16x32_f16.
// A sched_barrier between the fragment ds_loads and the WMMA chain keeps all
// fragments live in distinct registers so the loads pipeline (staged dscnt
// waits) instead of each WMMA serializing on s_wait_dscnt 0.
// ---------------------------------------------------------------------------
template<bool B_HALF, bool C_HALF, bool HAS_BIAS>
__global__ __launch_bounds__(256)
void wmma_gemm(const float* __restrict__ A,
               const void*  __restrict__ Bsrc,
               void*        __restrict__ Cdst,
               const float* __restrict__ bias,
               int K,
               long bRowStride, long bBatchExtra,
               long cRowStride, long cBatchExtra)
{
    constexpr int ROWB = 40;                  // f16 per LDS row: 32 used + 8 pad (80B rows)
    __shared__ __align__(16) _Float16 As[128 * ROWB];
    __shared__ __align__(16) _Float16 Bs[64 * ROWB];

    const int  tid    = threadIdx.x;
    const int  lane   = tid & 31;
    const int  wm     = tid >> 5;             // wave id == M sub-tile (16 rows each)
    const int  half   = lane >> 4;
    const int  l16    = lane & 15;
    const long nBlock = (long)blockIdx.x * 64;
    const int  mBlock = blockIdx.y * 128;
    const int  b      = (int)(nBlock / HW);   // 64-col tiles never cross a batch

    // A loader: row = tid/2 (128 rows), 16 consecutive K per thread (two K-groups)
    const int aRow = tid >> 1;
    const int aKH  = (tid & 1) * 16;                         // 0 or 16
    const int g0   = aKH >> 3;                               // group 0 or 2
    const int sgA0 = g0;                                     // frag layout {0-15 | 16-31}
    const int g1   = g0 + 1;
    const int sgA1 = g1;
    // B loader: k = tid/8 (32 rows), 8 consecutive n per thread; natural K order in LDS
    const int bK  = tid >> 3;
    const int bN0 = (tid & 7) * 8;

    // hoisted global pointers (advance per K-step)
    const float* aSrc = A + (long)(mBlock + aRow) * K + aKH;
    long bBase = (long)bK * bRowStride + nBlock + bN0 + (long)b * bBatchExtra;
    const long bStep = 32L * bRowStride;

    v8f acc0 = {}, acc1 = {}, acc2 = {}, acc3 = {};

    for (int k0 = 0; k0 < K; k0 += 32) {
        // ---- stage A tile (f32 -> f16, natural K order) ----
        {
            const float4* s4 = (const float4*)aSrc;
            float4 p0 = s4[0], p1 = s4[1], p2 = s4[2], p3 = s4[3];
            v8h hLo, hHi;
            hLo[0] = (_Float16)p0.x; hLo[1] = (_Float16)p0.y;
            hLo[2] = (_Float16)p0.z; hLo[3] = (_Float16)p0.w;
            hLo[4] = (_Float16)p1.x; hLo[5] = (_Float16)p1.y;
            hLo[6] = (_Float16)p1.z; hLo[7] = (_Float16)p1.w;
            hHi[0] = (_Float16)p2.x; hHi[1] = (_Float16)p2.y;
            hHi[2] = (_Float16)p2.z; hHi[3] = (_Float16)p2.w;
            hHi[4] = (_Float16)p3.x; hHi[5] = (_Float16)p3.y;
            hHi[6] = (_Float16)p3.z; hHi[7] = (_Float16)p3.w;
            *(v8h*)&As[aRow * ROWB + sgA0 * 8] = hLo;
            *(v8h*)&As[aRow * ROWB + sgA1 * 8] = hHi;
            aSrc += 32;
        }
        // ---- stage B tile (transpose to [n][k], f16) ----
        {
            _Float16 tmp[8];
            if (B_HALF) {
                v8h hv = *(const v8h*)((const _Float16*)Bsrc + bBase);
#pragma unroll
                for (int i = 0; i < 8; ++i) tmp[i] = hv[i];
            } else {
                const float4* s4 = (const float4*)((const float*)Bsrc + bBase);
                float4 p0 = s4[0], p1 = s4[1];
                tmp[0] = (_Float16)p0.x; tmp[1] = (_Float16)p0.y;
                tmp[2] = (_Float16)p0.z; tmp[3] = (_Float16)p0.w;
                tmp[4] = (_Float16)p1.x; tmp[5] = (_Float16)p1.y;
                tmp[6] = (_Float16)p1.z; tmp[7] = (_Float16)p1.w;
            }
#pragma unroll
            for (int i = 0; i < 8; ++i)
                Bs[(bN0 + i) * ROWB + bK] = tmp[i];
            bBase += bStep;
        }
        __syncthreads();

        // ---- preload ALL fragments (A + 4x B) ----
        const _Float16* ap  = &As[(wm * 16 + l16) * ROWB + half * 16];
        const _Float16* bp0 = &Bs[l16 * ROWB + half * 16];
        const _Float16* bp1 = bp0 + 16 * ROWB;
        const _Float16* bp2 = bp0 + 32 * ROWB;
        const _Float16* bp3 = bp0 + 48 * ROWB;

        v8h alo  = *(const v8h*)ap;        v8h ahi  = *(const v8h*)(ap + 8);
        v8h b0lo = *(const v8h*)bp0;       v8h b0hi = *(const v8h*)(bp0 + 8);
        v8h b1lo = *(const v8h*)bp1;       v8h b1hi = *(const v8h*)(bp1 + 8);
        v8h b2lo = *(const v8h*)bp2;       v8h b2hi = *(const v8h*)(bp2 + 8);
        v8h b3lo = *(const v8h*)bp3;       v8h b3hi = *(const v8h*)(bp3 + 8);

        v16h af  = V16(alo,  ahi);
        v16h bf0 = V16(b0lo, b0hi);
        v16h bf1 = V16(b1lo, b1hi);
        v16h bf2 = V16(b2lo, b2hi);
        v16h bf3 = V16(b3lo, b3hi);

        // keep every fragment live across this point -> distinct registers,
        // pipelined ds_loads, staged dscnt waits before the WMMA chain
        __builtin_amdgcn_sched_barrier(0);

        // ---- 4 back-to-back WMMAs, A fragment reused ----
        acc0 = __builtin_amdgcn_wmma_f32_16x16x32_f16(false, af, false, bf0,
                                                      (short)0, acc0, false, false);
        acc1 = __builtin_amdgcn_wmma_f32_16x16x32_f16(false, af, false, bf1,
                                                      (short)0, acc1, false, false);
        acc2 = __builtin_amdgcn_wmma_f32_16x16x32_f16(false, af, false, bf2,
                                                      (short)0, acc2, false, false);
        acc3 = __builtin_amdgcn_wmma_f32_16x16x32_f16(false, af, false, bf3,
                                                      (short)0, acc3, false, false);
        __syncthreads();
    }

    // ---- writeback: acc[s][r] -> (M = r + half*8, N = s*16 + l16) ----
    const long nOut = nBlock + l16;
    long rowAddr = (long)(mBlock + wm * 16 + half * 8) * cRowStride + nOut
                 + (long)b * cBatchExtra;
    const int mBase = mBlock + wm * 16 + half * 8;
#pragma unroll
    for (int r = 0; r < 8; ++r) {
        const float bv = HAS_BIAS ? bias[mBase + r] : 0.0f;
        const float v0 = acc0[r] + bv;
        const float v1 = acc1[r] + bv;
        const float v2 = acc2[r] + bv;
        const float v3 = acc3[r] + bv;
        if (C_HALF) {
            _Float16* cp = (_Float16*)Cdst + rowAddr;
            cp[0] = (_Float16)v0; cp[16] = (_Float16)v1;
            cp[32] = (_Float16)v2; cp[48] = (_Float16)v3;
        } else {
            float* cp = (float*)Cdst + rowAddr;
            cp[0] = v0; cp[16] = v1; cp[32] = v2; cp[48] = v3;
        }
        rowAddr += cRowStride;
    }
}

// ---------------------------------------------------------------------------
// 3x3 neighborhood attention, softmax over 9 taps (zero logit on padded taps).
// q:  [INNER  x NPIX] f16 (row = head*64+d)
// kv: [2*INNER x NPIX] f16 (k rows 0..511, v rows 512..1023)
// o:  [INNER  x NPIX] f16
// One thread per (head, pixel).
// ---------------------------------------------------------------------------
__global__ __launch_bounds__(256)
void natt_kernel(const _Float16* __restrict__ qb,
                 const _Float16* __restrict__ kvb,
                 _Float16*       __restrict__ ob)
{
    const int idx  = blockIdx.x * 256 + threadIdx.x;   // 0 .. HEADS*NPIX-1 (exact)
    const int head = idx / NPIX;
    const int n    = idx - head * NPIX;
    const int p    = n % HW;
    const int y    = p / WID;
    const int x    = p - y * WID;

    float msk[9];
    int   offc[9];
    {
        int t = 0;
#pragma unroll
        for (int dy = -1; dy <= 1; ++dy)
#pragma unroll
            for (int dx = -1; dx <= 1; ++dx, ++t) {
                const int yy = y + dy, xx = x + dx;
                const bool ok = (yy >= 0) & (yy < HGT) & (xx >= 0) & (xx < WID);
                msk[t]  = ok ? 1.0f : 0.0f;
                offc[t] = ok ? (dy * WID + dx) : 0;   // safe in-range address when masked
            }
    }

    const long qbase = (long)head * DH * NPIX + n;
    const long vbase = qbase + (long)INNER * NPIX;

    float dot[9];
#pragma unroll
    for (int t = 0; t < 9; ++t) dot[t] = 0.0f;

    for (int d = 0; d < DH; ++d) {
        const float qd = (float)qb[qbase + (long)d * NPIX];
        const long  kb = qbase + (long)d * NPIX;
#pragma unroll
        for (int t = 0; t < 9; ++t)
            dot[t] += qd * (float)kvb[kb + offc[t]];
    }

    float mx = 0.0f;   // padded taps contribute logit exactly 0
#pragma unroll
    for (int t = 0; t < 9; ++t) {
        dot[t] = msk[t] * dot[t] * ATT_SCALE;
        mx = fmaxf(mx, dot[t]);
    }
    float e[9];
    float s = 0.0f;
#pragma unroll
    for (int t = 0; t < 9; ++t) { e[t] = __expf(dot[t] - mx); s += e[t]; }
    const float inv = 1.0f / s;
#pragma unroll
    for (int t = 0; t < 9; ++t) e[t] = e[t] * inv * msk[t];   // padded v == 0

    for (int d = 0; d < DH; ++d) {
        const long vb = vbase + (long)d * NPIX;
        float o = 0.0f;
#pragma unroll
        for (int t = 0; t < 9; ++t)
            o += e[t] * (float)kvb[vb + offc[t]];
        ob[qbase + (long)d * NPIX] = (_Float16)o;
    }
}

// ---------------------------------------------------------------------------
extern "C" void kernel_launch(void* const* d_in, const int* in_sizes, int n_in,
                              void* d_out, int out_size, void* d_ws, size_t ws_size,
                              hipStream_t stream)
{
    const float* x   = (const float*)d_in[0];   // (4, 384, 56, 56)
    const float* Wq  = (const float*)d_in[1];   // (512, 384)
    const float* Wkv = (const float*)d_in[2];   // (1024, 384)
    const float* Wo  = (const float*)d_in[3];   // (384, 512)
    const float* bo  = (const float*)d_in[4];   // (384,)
    float* out = (float*)d_out;                 // (4, 384, 56, 56)

    // workspace partition (f16 intermediates, channel-major [m][NPIX])
    _Float16* q_buf  = (_Float16*)d_ws;                             // 512  x NPIX
    _Float16* kv_buf = q_buf  + (size_t)INNER * NPIX;               // 1024 x NPIX
    _Float16* ao_buf = kv_buf + (size_t)2 * INNER * NPIX;           // 512  x NPIX

    const long xExtra   = (long)HW * (DIM_C - 1);  // x batch hop inside B-matrix view
    const long outExtra = (long)HW * (DIM_C - 1);  // output batch hop

    // Q = Wq * X        (512 x 12544)
    wmma_gemm<false, true, false><<<dim3(NPIX / 64, INNER / 128), 256, 0, stream>>>(
        Wq, x, q_buf, nullptr, DIM_C, (long)HW, xExtra, (long)NPIX, 0L);

    // KV = Wkv * X      (1024 x 12544)
    wmma_gemm<false, true, false><<<dim3(NPIX / 64, (2 * INNER) / 128), 256, 0, stream>>>(
        Wkv, x, kv_buf, nullptr, DIM_C, (long)HW, xExtra, (long)NPIX, 0L);

    // neighborhood attention (one thread per head*pixel)
    natt_kernel<<<(HEADS * NPIX) / 256, 256, 0, stream>>>(q_buf, kv_buf, ao_buf);

    // OUT = Wo * AO + bo  (384 x 12544) -> (B, 384, H, W)
    wmma_gemm<true, false, true><<<dim3(NPIX / 64, DIM_C / 128), 256, 0, stream>>>(
        Wo, ao_buf, out, bo, INNER, (long)NPIX, 0L, (long)HW, outExtra);
}
